// att_temporal_50946902065525
// MI455X (gfx1250) — matrix operand order repaired and checked by reading, not compiled, not run on hardware
//
#include <hip/hip_runtime.h>
#include <hip/hip_bf16.h>

typedef __attribute__((ext_vector_type(4)))  float        fv4;
typedef __attribute__((ext_vector_type(4)))  unsigned int uv4;
typedef __attribute__((ext_vector_type(8)))  float        v8f;
typedef __attribute__((ext_vector_type(16))) __bf16       v16bf;

#define NB 8
#define NT 32
#define NN 256
#define NF 256
#define ND 256
#define SLAB 65536          // 256*256 elements per (b,t) slab
#define OUT0 16777216       // elements in "out" part of d_out (8*32*256*256)

union AFrag { v16bf v; uv4 q[2]; unsigned short u16[16]; };

__device__ __forceinline__ unsigned short f2bf(float f) {
    unsigned int u = __float_as_uint(f);
    unsigned int r = u + 0x7FFFu + ((u >> 16) & 1u);   // round-to-nearest-even
    return (unsigned short)(r >> 16);
}

__device__ __forceinline__ v8f vzero8() {
    v8f z = {0.f,0.f,0.f,0.f,0.f,0.f,0.f,0.f};
    return z;
}

// ---------------------------------------------------------------------------
// Kernel 1: row l2-normalize. One wave32 per row of F=256. Writes bf16 xn to
// workspace and pre-fills fp32 out[:,t] = xn[:,t] (t>=1) and out[:,0]=xn[:,1].
// ---------------------------------------------------------------------------
__global__ __launch_bounds__(256) void k_norm(const float* __restrict__ x,
                                              unsigned short* __restrict__ xnb,
                                              float* __restrict__ outBase) {
    const int g    = blockIdx.x * 8 + (threadIdx.x >> 5);   // row id, 0..65535
    const int lane = threadIdx.x & 31;
    const float* xp = x + (size_t)g * 256 + lane * 8;
    fv4 v0 = *(const fv4*)xp;
    fv4 v1 = *(const fv4*)(xp + 4);
    float s = v0[0]*v0[0] + v0[1]*v0[1] + v0[2]*v0[2] + v0[3]*v0[3]
            + v1[0]*v1[0] + v1[1]*v1[1] + v1[2]*v1[2] + v1[3]*v1[3];
    #pragma unroll
    for (int m = 16; m >= 1; m >>= 1) s += __shfl_xor(s, m, 32);
    const float rs = rsqrtf(fmaxf(s, 1e-12f));
    fv4 n0 = v0 * rs, n1 = v1 * rs;

    union { uv4 q; unsigned short u[8]; } pk;
    #pragma unroll
    for (int i = 0; i < 4; ++i) { pk.u[i] = f2bf(n0[i]); pk.u[4+i] = f2bf(n1[i]); }
    *(uv4*)(xnb + (size_t)g * 256 + lane * 8) = pk.q;

    const int n = g & 255, t = (g >> 8) & 31, bb = g >> 13;
    if (t >= 1) {
        float* op = outBase + (size_t)(bb * 32 + t) * SLAB + n * 256 + lane * 8;
        *(fv4*)op = n0; *(fv4*)(op + 4) = n1;
        if (t == 1) {
            float* o0 = outBase + (size_t)(bb * 32) * SLAB + n * 256 + lane * 8;
            *(fv4*)o0 = n0; *(fv4*)(o0 + 4) = n1;
        }
    }
}

// ---------------------------------------------------------------------------
// Kernel 2: transpose Wq/Wk to bf16 [d][f] (B-operand wants K-contiguous cols)
// ---------------------------------------------------------------------------
__global__ __launch_bounds__(256) void k_wt(const float* __restrict__ Wq,
                                            const float* __restrict__ Wk,
                                            unsigned short* __restrict__ wqT,
                                            unsigned short* __restrict__ wkT) {
    const int d = blockIdx.x, f = threadIdx.x;
    const float* W = blockIdx.y ? Wk : Wq;
    unsigned short* o = blockIdx.y ? wkT : wqT;
    o[d * 256 + f] = f2bf(W[f * 256 + d]);
}

// ---------------------------------------------------------------------------
// Kernel 3: projections  q = xn@Wq, k = xn@Wk  (bf16 WMMA, f32 accum).
// grid = (4 row-tiles, 256 slabs, 2 {q,k}); 8 waves, wave owns 16 rows x 128 cols.
// ---------------------------------------------------------------------------
__global__ __launch_bounds__(256) void k_proj(const unsigned short* __restrict__ xnb,
                                              const unsigned short* __restrict__ wqT,
                                              const unsigned short* __restrict__ wkT,
                                              unsigned short* __restrict__ qA,
                                              unsigned short* __restrict__ kA) {
    const int slab = blockIdx.y;
    const unsigned short* A  = xnb + (size_t)slab * SLAB;
    const unsigned short* Bm = blockIdx.z ? wkT : wqT;
    unsigned short* Out      = (blockIdx.z ? kA : qA) + (size_t)slab * SLAB;

    const int w = threadIdx.x >> 5, lane = threadIdx.x & 31;
    const int rb = w & 3, cb0 = (w >> 2) * 8;
    const int rowbase = blockIdx.x * 64 + rb * 16;
    const int hs = (lane < 16) ? 0 : 1;
    const int lrow = rowbase + (lane & 15);

    v8f acc[8];
    #pragma unroll
    for (int c = 0; c < 8; ++c) acc[c] = vzero8();

    for (int k0 = 0; k0 < NF; k0 += 32) {
        AFrag a;
        const unsigned short* ap = A + lrow * 256 + k0 + hs * 8;
        a.q[0] = *(const uv4*)ap;
        a.q[1] = *(const uv4*)(ap + 16);
        #pragma unroll
        for (int c = 0; c < 8; ++c) {
            const int col = (cb0 + c) * 16 + (lane & 15);
            AFrag bf;
            const unsigned short* bp = Bm + col * 256 + k0 + hs * 8;
            bf.q[0] = *(const uv4*)bp;
            bf.q[1] = *(const uv4*)(bp + 16);
            acc[c] = __builtin_amdgcn_wmma_f32_16x16x32_bf16(
                false, a.v, false, bf.v, (short)0, acc[c], false, false);
        }
    }
    #pragma unroll
    for (int c = 0; c < 8; ++c) {
        const int col = (cb0 + c) * 16 + (lane & 15);
        #pragma unroll
        for (int v = 0; v < 8; ++v)
            Out[(rowbase + hs * 8 + v) * 256 + col] = f2bf(acc[c][v]);
    }
}

// ---------------------------------------------------------------------------
// Kernel 4: att = softmax(q k^T / 10).  k row-major IS the B operand layout.
// 64x256 score tile per block; fused row softmax; fp32 att -> d_out att region.
// ---------------------------------------------------------------------------
__global__ __launch_bounds__(256) void k_att(const unsigned short* __restrict__ qA,
                                             const unsigned short* __restrict__ kA,
                                             float* __restrict__ attOut) {
    const int y = blockIdx.y;               // 0..247
    const int bb = y / 31, t = y % 31;
    const unsigned short* Q = qA + (size_t)(bb * 32 + t + 1) * SLAB;
    const unsigned short* K = kA + (size_t)(bb * 32 + t) * SLAB;
    float* att = attOut + (size_t)y * SLAB;

    const int w = threadIdx.x >> 5, lane = threadIdx.x & 31;
    const int rb = w & 3, ch = w >> 2, cb0 = ch * 8;
    const int rowbase = blockIdx.x * 64 + rb * 16;
    const int hs = (lane < 16) ? 0 : 1;
    const int lrow = rowbase + (lane & 15);

    v8f acc[8];
    #pragma unroll
    for (int c = 0; c < 8; ++c) acc[c] = vzero8();

    for (int d0 = 0; d0 < ND; d0 += 32) {
        AFrag a;
        const unsigned short* ap = Q + lrow * 256 + d0 + hs * 8;
        a.q[0] = *(const uv4*)ap;
        a.q[1] = *(const uv4*)(ap + 16);
        #pragma unroll
        for (int c = 0; c < 8; ++c) {
            const int j = (cb0 + c) * 16 + (lane & 15);
            AFrag bf;
            const unsigned short* bp = K + j * 256 + d0 + hs * 8;
            bf.q[0] = *(const uv4*)bp;
            bf.q[1] = *(const uv4*)(bp + 16);
            acc[c] = __builtin_amdgcn_wmma_f32_16x16x32_bf16(
                false, a.v, false, bf.v, (short)0, acc[c], false, false);
        }
    }
    #pragma unroll
    for (int c = 0; c < 8; ++c)
        #pragma unroll
        for (int v = 0; v < 8; ++v) acc[c][v] *= 0.1f;

    __shared__ float mbuf[2][64];
    __shared__ float sbuf[2][64];

    float lm[8];
    #pragma unroll
    for (int v = 0; v < 8; ++v) {
        float m = acc[0][v];
        #pragma unroll
        for (int c = 1; c < 8; ++c) m = fmaxf(m, acc[c][v]);
        #pragma unroll
        for (int msk = 8; msk >= 1; msk >>= 1) m = fmaxf(m, __shfl_xor(m, msk, 32));
        lm[v] = m;
    }
    if ((lane & 15) == 0) {
        #pragma unroll
        for (int v = 0; v < 8; ++v) mbuf[ch][rb * 16 + hs * 8 + v] = lm[v];
    }
    __syncthreads();

    float ls[8];
    #pragma unroll
    for (int v = 0; v < 8; ++v) {
        const int rl = rb * 16 + hs * 8 + v;
        const float m = fmaxf(mbuf[0][rl], mbuf[1][rl]);
        float s = 0.f;
        #pragma unroll
        for (int c = 0; c < 8; ++c) {
            const float e = __expf(acc[c][v] - m);
            acc[c][v] = e;
            s += e;
        }
        #pragma unroll
        for (int msk = 8; msk >= 1; msk >>= 1) s += __shfl_xor(s, msk, 32);
        ls[v] = s;
    }
    if ((lane & 15) == 0) {
        #pragma unroll
        for (int v = 0; v < 8; ++v) sbuf[ch][rb * 16 + hs * 8 + v] = ls[v];
    }
    __syncthreads();

    #pragma unroll
    for (int v = 0; v < 8; ++v) {
        const int rl = rb * 16 + hs * 8 + v;
        const float inv = 1.0f / (sbuf[0][rl] + sbuf[1][rl]);
        const int i = rowbase + hs * 8 + v;
        #pragma unroll
        for (int c = 0; c < 8; ++c)
            att[i * 256 + (cb0 + c) * 16 + (lane & 15)] = acc[c][v] * inv;
    }
}

// ---------------------------------------------------------------------------
// Kernel 5: prog = att @ prev ; out += prog (out was pre-filled with xn).
// prev staged through LDS transposed ([f][j], stride 40 -> 16B aligned,
// conflict-free b128 reads) so the B operand is a contiguous LDS load.
// ---------------------------------------------------------------------------
__global__ __launch_bounds__(256) void k_prog(const unsigned short* __restrict__ xnb,
                                              const float* __restrict__ attIn,
                                              float* __restrict__ outBase) {
    const int y = blockIdx.y;
    const int bb = y / 31, t = y % 31;
    const unsigned short* P = xnb + (size_t)(bb * 32 + t) * SLAB;   // prev
    const float* att = attIn + (size_t)y * SLAB;
    float* out = outBase + (size_t)(bb * 32 + t + 1) * SLAB;

    __shared__ unsigned short prevT[256][40];   // [f][j-local], 20 KB

    const int w = threadIdx.x >> 5, lane = threadIdx.x & 31;
    const int rb = w & 3, cb0 = (w >> 2) * 8;
    const int rowbase = blockIdx.x * 64 + rb * 16;
    const int hs = (lane < 16) ? 0 : 1;
    const int lrow = rowbase + (lane & 15);

    v8f acc[8];
    #pragma unroll
    for (int c = 0; c < 8; ++c) acc[c] = vzero8();

    for (int j0 = 0; j0 < NN; j0 += 32) {
        // stage prev[j0..j0+31][0..255] transposed into LDS
        {
            const int jj = threadIdx.x >> 3;          // 0..31
            const int fb = (threadIdx.x & 7) * 32;    // 0..224
            const unsigned short* src = P + (j0 + jj) * 256 + fb;
            union { uv4 q[4]; unsigned short s[32]; } ld;
            ld.q[0] = *(const uv4*)(src);
            ld.q[1] = *(const uv4*)(src + 8);
            ld.q[2] = *(const uv4*)(src + 16);
            ld.q[3] = *(const uv4*)(src + 24);
            #pragma unroll
            for (int e = 0; e < 32; ++e) prevT[fb + e][jj] = ld.s[e];
        }
        __syncthreads();

        // A operand: att row, fp32 -> bf16
        AFrag a;
        {
            const float* ap = att + lrow * 256 + j0 + hs * 8;
            fv4 t0 = *(const fv4*)ap;
            fv4 t1 = *(const fv4*)(ap + 4);
            fv4 t2 = *(const fv4*)(ap + 16);
            fv4 t3 = *(const fv4*)(ap + 20);
            #pragma unroll
            for (int i = 0; i < 4; ++i) {
                a.u16[i]      = f2bf(t0[i]);
                a.u16[4 + i]  = f2bf(t1[i]);
                a.u16[8 + i]  = f2bf(t2[i]);
                a.u16[12 + i] = f2bf(t3[i]);
            }
        }
        #pragma unroll
        for (int c = 0; c < 8; ++c) {
            const int f = (cb0 + c) * 16 + (lane & 15);
            AFrag bf;
            const unsigned short* bp = &prevT[f][hs * 8];
            bf.q[0] = *(const uv4*)bp;
            bf.q[1] = *(const uv4*)(bp + 16);
            acc[c] = __builtin_amdgcn_wmma_f32_16x16x32_bf16(
                false, a.v, false, bf.v, (short)0, acc[c], false, false);
        }
        __syncthreads();
    }

    #pragma unroll
    for (int c = 0; c < 8; ++c) {
        const int f = (cb0 + c) * 16 + (lane & 15);
        #pragma unroll
        for (int v = 0; v < 8; ++v) {
            const int idx = (rowbase + hs * 8 + v) * 256 + f;
            out[idx] = out[idx] + acc[c][v];
        }
    }
}

// ---------------------------------------------------------------------------
extern "C" void kernel_launch(void* const* d_in, const int* in_sizes, int n_in,
                              void* d_out, int out_size, void* d_ws, size_t ws_size,
                              hipStream_t stream) {
    const float* x  = (const float*)d_in[0];
    const float* Wq = (const float*)d_in[1];
    const float* Wk = (const float*)d_in[2];
    float* out = (float*)d_out;
    float* att = out + OUT0;

    unsigned short* ws  = (unsigned short*)d_ws;
    unsigned short* xnb = ws;                       // 16,777,216 bf16
    unsigned short* qA  = ws + 16777216;            // 16,777,216 bf16
    unsigned short* kA  = ws + 2 * 16777216;        // 16,777,216 bf16
    unsigned short* wqT = ws + 3 * 16777216;        // 65,536 bf16
    unsigned short* wkT = wqT + 65536;              // 65,536 bf16

    k_norm<<<8192, 256, 0, stream>>>(x, xnb, out);
    k_wt  <<<dim3(256, 2), 256, 0, stream>>>(Wq, Wk, wqT, wkT);
    k_proj<<<dim3(4, 256, 2), 256, 0, stream>>>(xnb, wqT, wkT, qA, kA);
    k_att <<<dim3(4, 248), 256, 0, stream>>>(qA, kA, att);
    k_prog<<<dim3(4, 248), 256, 0, stream>>>(xnb, att, out);
}
